// MF_Transformer_86079734547179
// MI455X (gfx1250) — compile-verified, matrix operand order checked by reading, not run on hardware
//
#include <hip/hip_runtime.h>
#include <hip/hip_bf16.h>

// Problem constants
#define BB    128
#define SS    50
#define DDIM  1024
#define DFFC  4096
#define HEADS 16
#define DK    64
#define LAYERS 5
#define NCLS  71
#define KRAW  1200
#define KPAD  1216              // 1200 padded to multiple of 32
#define MROWS (BB*SS)           // 6400
#define NQKV  (3*DDIM)          // 3072

typedef __attribute__((ext_vector_type(16))) __bf16 v16bf;
typedef __attribute__((ext_vector_type(8)))  float  v8f;
typedef int async_i4 __attribute__((vector_size(16)));   // int4 for async builtins

#if defined(__has_builtin)
#if __has_builtin(__builtin_amdgcn_global_load_async_to_lds_b128) && \
    __has_builtin(__builtin_amdgcn_s_wait_asynccnt)
#define HAVE_ASYNC 1
#endif
#endif
#ifndef HAVE_ASYNC
#define HAVE_ASYNC 0
#endif

__device__ __forceinline__ unsigned short f2bf(float f) {
  unsigned int u = __float_as_uint(f);
  u += 0x7FFFu + ((u >> 16) & 1u);           // round-to-nearest-even
  return (unsigned short)(u >> 16);
}
__device__ __forceinline__ float bf2f(unsigned short h) {
  return __uint_as_float(((unsigned int)h) << 16);
}

// ---------------- pack / convert kernels ----------------
// generic transpose-convert: src f32 [K][N] -> dst bf16 [N][K]
__global__ void convert_t_kernel(const float* __restrict__ src,
                                 unsigned short* __restrict__ dst, int K, int N) {
  long i = (long)blockIdx.x * blockDim.x + threadIdx.x;
  long n = (long)K * N;
  if (i >= n) return;
  int k = (int)(i / N), c = (int)(i % N);
  dst[(long)c * K + k] = f2bf(src[i]);
}

// x (128,5000,12) viewed as [6400][1200] -> bf16 [6400][1216] zero padded (A, row major)
__global__ void pack_x_kernel(const float* __restrict__ x, unsigned short* __restrict__ xp) {
  long i = (long)blockIdx.x * blockDim.x + threadIdx.x;
  long n = (long)MROWS * KPAD;
  if (i >= n) return;
  int m = (int)(i / KPAD), c = (int)(i % KPAD);
  xp[i] = (c < KRAW) ? f2bf(x[(long)m * KRAW + c]) : 0;
}

// embed_w [1200][1024] -> Bt bf16 [1024][1216] zero padded
__global__ void pack_emb_kernel(const float* __restrict__ w, unsigned short* __restrict__ wp) {
  long i = (long)blockIdx.x * blockDim.x + threadIdx.x;
  long n = (long)DDIM * KPAD;
  if (i >= n) return;
  int e = (int)(i / KPAD), k = (int)(i % KPAD);
  wp[i] = (k < KRAW) ? f2bf(w[(long)k * DDIM + e]) : 0;
}

// attn_w [L,4,D,D]: projections 0..2 -> per layer Bt[(p*D+e)][k]
__global__ void pack_qkvw_kernel(const float* __restrict__ aw, unsigned short* __restrict__ dst) {
  long i = (long)blockIdx.x * blockDim.x + threadIdx.x;
  long n = (long)LAYERS * 3 * DDIM * DDIM;
  if (i >= n) return;
  int l = (int)(i / (3L * DDIM * DDIM));
  long r = i % (3L * DDIM * DDIM);
  int p = (int)(r / ((long)DDIM * DDIM));
  long r2 = r % ((long)DDIM * DDIM);
  int k = (int)(r2 / DDIM), e = (int)(r2 % DDIM);
  dst[(long)l * DDIM * NQKV + (long)(p * DDIM + e) * DDIM + k] =
      f2bf(aw[(long)l * 4 * DDIM * DDIM + (long)p * DDIM * DDIM + (long)k * DDIM + e]);
}

// ---------------- generic bf16 WMMA GEMM (B pre-transposed) ----------------
// C[M,N] = act(A[M,K] @ Bt[N,K]^T + bias) (+resid); outputs f32 and/or bf16.
// M % 64 == 0, N % 256 == 0, K % 32 == 0.
// Block tile 64x256, 8 waves, wave tile 32x64 (2x4 WMMA accumulators), BK=32,
// double-buffered LDS staged with async global->LDS copies (ASYNCcnt).

__device__ __forceinline__ void stage_tile(const unsigned short* __restrict__ A,
                                           const unsigned short* __restrict__ Bt,
                                           unsigned short* lsA, unsigned short* lsB,
                                           int m0, int n0, int K, int k0) {
  const int t = (int)threadIdx.x;
  const unsigned short* gsrc[5];
  unsigned short* ldst[5];
  {
    int arow = t >> 2, achk = t & 3;                   // 64 rows x 4 chunks
    gsrc[0] = A + (size_t)(m0 + arow) * K + k0 + achk * 8;
    ldst[0] = lsA + arow * 32 + achk * 8;
  }
#pragma unroll
  for (int it = 0; it < 4; ++it) {                     // 256 cols x 4 chunks
    int chunk = t + it * 256;
    int nrow = chunk >> 2, chk = chunk & 3;
    gsrc[1 + it] = Bt + (size_t)(n0 + nrow) * K + k0 + chk * 8;
    ldst[1 + it] = lsB + nrow * 32 + chk * 8;
  }
#if HAVE_ASYNC
#pragma unroll
  for (int i = 0; i < 5; ++i)
    __builtin_amdgcn_global_load_async_to_lds_b128(
        (__attribute__((address_space(1))) async_i4*)gsrc[i],
        (__attribute__((address_space(3))) async_i4*)ldst[i], 0, 0);
#else
  uint4 r[5];
#pragma unroll
  for (int i = 0; i < 5; ++i) r[i] = *reinterpret_cast<const uint4*>(gsrc[i]);
#pragma unroll
  for (int i = 0; i < 5; ++i) *reinterpret_cast<uint4*>(ldst[i]) = r[i];
#endif
}

__global__ __launch_bounds__(256) void wmma_gemm_kernel(
    const unsigned short* __restrict__ A, const unsigned short* __restrict__ Bt,
    const float* __restrict__ bias, const float* __restrict__ resid,
    float* __restrict__ outF, unsigned short* __restrict__ outB,
    int M, int N, int K, int relu) {
  __shared__ __align__(16) unsigned short lsA[2][64 * 32];    // [row][k]
  __shared__ __align__(16) unsigned short lsB[2][256 * 32];   // [col][k]

  const int t    = (int)threadIdx.x;
  const int lane = t & 31;
  const int wave = t >> 5;
  const int wm   = wave >> 2;        // 0..1  (M direction, 32 rows each)
  const int wn   = wave & 3;         // 0..3  (N direction, 64 cols each)
  const int hsel = (lane < 16) ? 0 : 8;
  const int rr   = lane & 15;
  const int n0   = blockIdx.x * 256;
  const int m0   = blockIdx.y * 64;

  v8f acc[2][4] = {};
  const int nk = K >> 5;

  stage_tile(A, Bt, lsA[0], lsB[0], m0, n0, K, 0);

  for (int ks = 0; ks < nk; ++ks) {
    const int cur = ks & 1;
    if (ks + 1 < nk) {
      stage_tile(A, Bt, lsA[cur ^ 1], lsB[cur ^ 1], m0, n0, K, (ks + 1) * 32);
#if HAVE_ASYNC
      __builtin_amdgcn_s_wait_asynccnt(5);   // previous tile's 5 copies done
#endif
    } else {
#if HAVE_ASYNC
      __builtin_amdgcn_s_wait_asynccnt(0);
#endif
    }
    __syncthreads();

    // ---- fragments per CDNA5 16-bit A/B VGPR layout (K-pairs contiguous) ----
    union Frag { v16bf v; uint4 q[2]; };
    Frag af[2], bfr[4];
    const unsigned short* pa = lsA[cur];
    const unsigned short* pb = lsB[cur];
#pragma unroll
    for (int mi = 0; mi < 2; ++mi) {
      int base = (wm * 32 + mi * 16 + rr) * 32 + hsel;
      af[mi].q[0] = *reinterpret_cast<const uint4*>(&pa[base]);
      af[mi].q[1] = *reinterpret_cast<const uint4*>(&pa[base + 16]);
    }
#pragma unroll
    for (int ni = 0; ni < 4; ++ni) {
      int base = (wn * 64 + ni * 16 + rr) * 32 + hsel;
      bfr[ni].q[0] = *reinterpret_cast<const uint4*>(&pb[base]);
      bfr[ni].q[1] = *reinterpret_cast<const uint4*>(&pb[base + 16]);
    }
#pragma unroll
    for (int mi = 0; mi < 2; ++mi)
#pragma unroll
      for (int ni = 0; ni < 4; ++ni)
        acc[mi][ni] = __builtin_amdgcn_wmma_f32_16x16x32_bf16(
            false, af[mi].v, false, bfr[ni].v, (short)0, acc[mi][ni], false, false);
    __syncthreads();
  }

  // ---- epilogue: D layout VGPR r, lane t: N=t%16, M=r+8*(t/16) ----
#pragma unroll
  for (int mi = 0; mi < 2; ++mi)
#pragma unroll
    for (int ni = 0; ni < 4; ++ni) {
      int n = n0 + wn * 64 + ni * 16 + rr;
#pragma unroll
      for (int r = 0; r < 8; ++r) {
        int m = m0 + wm * 32 + mi * 16 + hsel + r;
        float v = acc[mi][ni][r];
        if (bias)  v += bias[n];
        if (relu)  v = fmaxf(v, 0.0f);
        if (resid) v += resid[(size_t)m * N + n];
        if (outF)  outF[(size_t)m * N + n] = v;
        if (outB)  outB[(size_t)m * N + n] = f2bf(v);
      }
    }
}

// ---------------- positional encoding add (in-place on f32 h) ----------------
__global__ void pe_add_kernel(float* __restrict__ h) {
  long i = (long)blockIdx.x * blockDim.x + threadIdx.x;
  long n = (long)MROWS * DDIM;
  if (i >= n) return;
  int m = (int)(i / DDIM), d = (int)(i % DDIM);
  int s = m % SS;
  float freq = __expf(-(float)(d & ~1) * (9.210340371976184f / (float)DDIM));
  float ph = (float)s * freq;
  h[i] += (d & 1) ? __cosf(ph) : __sinf(ph);
}

// ---------------- LayerNorm (ddof=1, /(std+1e-6)) f32 -> bf16 ----------------
__global__ __launch_bounds__(256) void ln_kernel(const float* __restrict__ x,
                                                 const float* __restrict__ g,
                                                 const float* __restrict__ b,
                                                 unsigned short* __restrict__ out) {
  __shared__ float red[256];
  const int row = blockIdx.x, t = threadIdx.x;
  const float* xr = x + (size_t)row * DDIM;
  float s = 0.0f;
  for (int d = t; d < DDIM; d += 256) s += xr[d];
  red[t] = s; __syncthreads();
  for (int st = 128; st > 0; st >>= 1) { if (t < st) red[t] += red[t + st]; __syncthreads(); }
  float mean = red[0] / (float)DDIM;
  __syncthreads();
  float ss = 0.0f;
  for (int d = t; d < DDIM; d += 256) { float dv = xr[d] - mean; ss += dv * dv; }
  red[t] = ss; __syncthreads();
  for (int st = 128; st > 0; st >>= 1) { if (t < st) red[t] += red[t + st]; __syncthreads(); }
  float fac = 1.0f / (sqrtf(red[0] / (float)(DDIM - 1)) + 1e-6f);
  for (int d = t; d < DDIM; d += 256)
    out[(size_t)row * DDIM + d] = f2bf(g[d] * ((xr[d] - mean) * fac) + b[d]);
}

// ---------------- attention: one thread per (b, head, q) ----------------
__global__ __launch_bounds__(256) void attn_kernel(const unsigned short* __restrict__ qkv,
                                                   unsigned short* __restrict__ ao) {
  int tid = blockIdx.x * blockDim.x + threadIdx.x;
  if (tid >= BB * HEADS * SS) return;
  int b = tid / (HEADS * SS);
  int r = tid % (HEADS * SS);
  int hh = r / SS, q = r % SS;

  const unsigned short* qp = qkv + (size_t)(b * SS + q) * NQKV + hh * DK;
  float qv[DK];
#pragma unroll
  for (int d = 0; d < DK; ++d) qv[d] = bf2f(qp[d]);

  float sc[SS];
  float mx = -1e30f;
  for (int kk = 0; kk < SS; ++kk) {
    const unsigned short* kp = qkv + (size_t)(b * SS + kk) * NQKV + DDIM + hh * DK;
    float s = 0.0f;
#pragma unroll
    for (int d = 0; d < DK; ++d) s += qv[d] * bf2f(kp[d]);
    s *= 0.125f;                                     // 1/sqrt(64)
    sc[kk] = s;
    mx = fmaxf(mx, s);
  }
  float sum = 0.0f;
  for (int kk = 0; kk < SS; ++kk) { float e = __expf(sc[kk] - mx); sc[kk] = e; sum += e; }
  float rinv = 1.0f / sum;

  float out[DK];
#pragma unroll
  for (int d = 0; d < DK; ++d) out[d] = 0.0f;
  for (int kk = 0; kk < SS; ++kk) {
    float p = sc[kk] * rinv;
    const unsigned short* vp = qkv + (size_t)(b * SS + kk) * NQKV + 2 * DDIM + hh * DK;
#pragma unroll
    for (int d = 0; d < DK; ++d) out[d] += p * bf2f(vp[d]);
  }
  unsigned short* op = ao + (size_t)(b * SS + q) * DDIM + hh * DK;
#pragma unroll
  for (int d = 0; d < DK; ++d) op[d] = f2bf(out[d]);
}

// ---------------- head: final LN of last row + cf BN scale -> bf16 ----------------
__global__ __launch_bounds__(256) void head_pool_kernel(const float* __restrict__ h,
                                                        const float* __restrict__ fg,
                                                        const float* __restrict__ fb,
                                                        const float* __restrict__ bng,
                                                        const float* __restrict__ bnb,
                                                        unsigned short* __restrict__ out) {
  __shared__ float red[256];
  const int b = blockIdx.x, t = threadIdx.x;
  const float* xr = h + (size_t)(b * SS + (SS - 1)) * DDIM;
  float s = 0.0f;
  for (int d = t; d < DDIM; d += 256) s += xr[d];
  red[t] = s; __syncthreads();
  for (int st = 128; st > 0; st >>= 1) { if (t < st) red[t] += red[t + st]; __syncthreads(); }
  float mean = red[0] / (float)DDIM;
  __syncthreads();
  float ss = 0.0f;
  for (int d = t; d < DDIM; d += 256) { float dv = xr[d] - mean; ss += dv * dv; }
  red[t] = ss; __syncthreads();
  for (int st = 128; st > 0; st >>= 1) { if (t < st) red[t] += red[t + st]; __syncthreads(); }
  float fac = 1.0f / (sqrtf(red[0] / (float)(DDIM - 1)) + 1e-6f);
  const float inv = rsqrtf(1.0f + 1e-5f);
  for (int d = t; d < DDIM; d += 256) {
    float lnv = fg[d] * ((xr[d] - mean) * fac) + fb[d];
    out[(size_t)b * DDIM + d] = f2bf(lnv * inv * bng[d] + bnb[d]);
  }
}

__global__ void head_scale_kernel(const float* __restrict__ z,
                                  const float* __restrict__ bng,
                                  const float* __restrict__ bnb,
                                  unsigned short* __restrict__ out) {
  int i = blockIdx.x * blockDim.x + threadIdx.x;
  if (i >= BB * DDIM) return;
  int d = i % DDIM;
  const float inv = rsqrtf(1.0f + 1e-5f);
  out[i] = f2bf(z[i] * inv * bng[d] + bnb[d]);
}

__global__ void final_fc_kernel(const unsigned short* __restrict__ z,
                                const float* __restrict__ w,
                                const float* __restrict__ bias,
                                float* __restrict__ out) {
  int tid = blockIdx.x * blockDim.x + threadIdx.x;
  if (tid >= BB * NCLS) return;
  int b = tid / NCLS, c = tid % NCLS;
  float s = bias[c];
  for (int d = 0; d < DDIM; ++d) s += bf2f(z[(size_t)b * DDIM + d]) * w[(size_t)d * NCLS + c];
  out[tid] = s;
}

// ---------------- host launcher ----------------
static inline unsigned gblk(long n) { return (unsigned)((n + 255) / 256); }

extern "C" void kernel_launch(void* const* d_in, const int* in_sizes, int n_in,
                              void* d_out, int out_size, void* d_ws, size_t ws_size,
                              hipStream_t stream) {
  (void)in_sizes; (void)n_in; (void)out_size; (void)ws_size;
  const float* x       = (const float*)d_in[0];
  const float* embed_w = (const float*)d_in[1];
  const float* attn_w  = (const float*)d_in[2];
  const float* attn_b  = (const float*)d_in[3];
  const float* ff_w1   = (const float*)d_in[4];
  const float* ff_b1   = (const float*)d_in[5];
  const float* ff_w2   = (const float*)d_in[6];
  const float* ff_b2   = (const float*)d_in[7];
  const float* ln_g    = (const float*)d_in[8];
  const float* ln_b    = (const float*)d_in[9];
  const float* fin_g   = (const float*)d_in[10];
  const float* fin_b   = (const float*)d_in[11];
  const float* cf_bn_g = (const float*)d_in[12];
  const float* cf_bn_b = (const float*)d_in[13];
  const float* cf_w    = (const float*)d_in[14];
  const float* cf_b    = (const float*)d_in[15];
  const float* fc_bn_g = (const float*)d_in[16];
  const float* fc_bn_b = (const float*)d_in[17];
  const float* fc_w    = (const float*)d_in[18];
  const float* fc_b    = (const float*)d_in[19];
  float* out = (float*)d_out;

  char* ws = (char*)d_ws;
  size_t off = 0;
  auto take = [&](size_t bytes) -> void* {
    void* p = ws + off;
    off += (bytes + 255) & ~(size_t)255;
    return p;
  };
  unsigned short* xp    = (unsigned short*)take((size_t)MROWS * KPAD * 2);
  unsigned short* embp  = (unsigned short*)take((size_t)DDIM * KPAD * 2);   // Bt
  unsigned short* qkvw  = (unsigned short*)take((size_t)LAYERS * NQKV * DDIM * 2);  // Bt
  unsigned short* ow    = (unsigned short*)take((size_t)LAYERS * DDIM * DDIM * 2);  // Bt
  unsigned short* w1b   = (unsigned short*)take((size_t)LAYERS * DFFC * DDIM * 2);  // Bt
  unsigned short* w2b   = (unsigned short*)take((size_t)LAYERS * DDIM * DFFC * 2);  // Bt
  unsigned short* cfwb  = (unsigned short*)take((size_t)DDIM * DDIM * 2);           // Bt
  float*          h     = (float*)take((size_t)MROWS * DDIM * 4);
  unsigned short* a     = (unsigned short*)take((size_t)MROWS * DDIM * 2);
  unsigned short* qkv   = (unsigned short*)take((size_t)MROWS * NQKV * 2);
  unsigned short* ao    = (unsigned short*)take((size_t)MROWS * DDIM * 2);
  unsigned short* mid   = (unsigned short*)take((size_t)MROWS * DFFC * 2);
  unsigned short* pooled= (unsigned short*)take((size_t)BB * DDIM * 2);
  float*          zf    = (float*)take((size_t)BB * DDIM * 4);
  unsigned short* z2    = (unsigned short*)take((size_t)BB * DDIM * 2);

  // pack weights & inputs (bf16, weights transposed to [N][K])
  pack_x_kernel<<<gblk((long)MROWS * KPAD), 256, 0, stream>>>(x, xp);
  pack_emb_kernel<<<gblk((long)DDIM * KPAD), 256, 0, stream>>>(embed_w, embp);
  pack_qkvw_kernel<<<gblk((long)LAYERS * 3 * DDIM * DDIM), 256, 0, stream>>>(attn_w, qkvw);
  for (int l = 0; l < LAYERS; ++l) {
    convert_t_kernel<<<gblk((long)DDIM * DDIM), 256, 0, stream>>>(
        attn_w + (size_t)l * 4 * DDIM * DDIM + 3 * DDIM * DDIM,
        ow + (size_t)l * DDIM * DDIM, DDIM, DDIM);
    convert_t_kernel<<<gblk((long)DDIM * DFFC), 256, 0, stream>>>(
        ff_w1 + (size_t)l * DDIM * DFFC, w1b + (size_t)l * DFFC * DDIM, DDIM, DFFC);
    convert_t_kernel<<<gblk((long)DFFC * DDIM), 256, 0, stream>>>(
        ff_w2 + (size_t)l * DFFC * DDIM, w2b + (size_t)l * DDIM * DFFC, DFFC, DDIM);
  }
  convert_t_kernel<<<gblk((long)DDIM * DDIM), 256, 0, stream>>>(cf_w, cfwb, DDIM, DDIM);

  // embed + positional encoding
  wmma_gemm_kernel<<<dim3(DDIM / 256, MROWS / 64), 256, 0, stream>>>(
      xp, embp, nullptr, nullptr, h, nullptr, MROWS, DDIM, KPAD, 0);
  pe_add_kernel<<<gblk((long)MROWS * DDIM), 256, 0, stream>>>(h);

  // transformer layers
  for (int l = 0; l < LAYERS; ++l) {
    ln_kernel<<<MROWS, 256, 0, stream>>>(h, ln_g + (size_t)(l * 2) * DDIM,
                                         ln_b + (size_t)(l * 2) * DDIM, a);
    wmma_gemm_kernel<<<dim3(NQKV / 256, MROWS / 64), 256, 0, stream>>>(
        a, qkvw + (size_t)l * NQKV * DDIM, attn_b + (size_t)l * 4 * DDIM, nullptr,
        nullptr, qkv, MROWS, NQKV, DDIM, 0);
    attn_kernel<<<gblk((long)BB * HEADS * SS), 256, 0, stream>>>(qkv, ao);
    wmma_gemm_kernel<<<dim3(DDIM / 256, MROWS / 64), 256, 0, stream>>>(
        ao, ow + (size_t)l * DDIM * DDIM, attn_b + (size_t)l * 4 * DDIM + 3 * DDIM, h,
        h, nullptr, MROWS, DDIM, DDIM, 0);
    ln_kernel<<<MROWS, 256, 0, stream>>>(h, ln_g + (size_t)(l * 2 + 1) * DDIM,
                                         ln_b + (size_t)(l * 2 + 1) * DDIM, a);
    wmma_gemm_kernel<<<dim3(DFFC / 256, MROWS / 64), 256, 0, stream>>>(
        a, w1b + (size_t)l * DFFC * DDIM, ff_b1 + (size_t)l * DFFC, nullptr,
        nullptr, mid, MROWS, DFFC, DDIM, 1);
    wmma_gemm_kernel<<<dim3(DDIM / 256, MROWS / 64), 256, 0, stream>>>(
        mid, w2b + (size_t)l * DDIM * DFFC, ff_b2 + (size_t)l * DDIM, h,
        h, nullptr, MROWS, DDIM, DFFC, 0);
  }

  // head
  head_pool_kernel<<<BB, 256, 0, stream>>>(h, fin_g, fin_b, cf_bn_g, cf_bn_b, pooled);
  wmma_gemm_kernel<<<dim3(DDIM / 256, BB / 64), 256, 0, stream>>>(
      pooled, cfwb, cf_b, nullptr, zf, nullptr, BB, DDIM, DDIM, 1);
  head_scale_kernel<<<gblk((long)BB * DDIM), 256, 0, stream>>>(zf, fc_bn_g, fc_bn_b, z2);
  final_fc_kernel<<<gblk((long)BB * NCLS), 256, 0, stream>>>(zf == nullptr ? nullptr : z2, fc_w, fc_b, out);
}
